// LigandProteinCrossAttention_75282186764811
// MI455X (gfx1250) — matrix operand order repaired and checked by reading, not compiled, not run on hardware
//
#include <hip/hip_runtime.h>
#include <hip/hip_bf16.h>
#include <math.h>

// ---------------------------------------------------------------------------
// Problem constants (from reference setup_inputs)
// ---------------------------------------------------------------------------
static constexpr int Bc  = 8;
static constexpr int Sc  = 2048;
static constexpr int Ac  = 128;
static constexpr int Hc  = 8;
static constexpr int Dc  = 256;   // model dim
static constexpr int PDc = 384;
static constexpr int LDc = 128;
// HD = Dc/Hc = 32

typedef __attribute__((ext_vector_type(16))) __bf16 v16bf;
typedef __attribute__((ext_vector_type(8)))  __bf16 v8bf;
typedef __attribute__((ext_vector_type(8)))  float  v8f;

static __device__ __forceinline__ v8f wmma_bf16(v16bf a, v16bf b, v8f c) {
  // D = A(16x32 bf16) * B(32x16 bf16) + C(16x16 f32)
  return __builtin_amdgcn_wmma_f32_16x16x32_bf16(false, a, false, b, (short)0, c,
                                                 false, false);
}

static __device__ __forceinline__ v8bf pack8(float4 a, float4 b) {
  v8bf r;
  r[0] = (__bf16)a.x; r[1] = (__bf16)a.y; r[2] = (__bf16)a.z; r[3] = (__bf16)a.w;
  r[4] = (__bf16)b.x; r[5] = (__bf16)b.y; r[6] = (__bf16)b.z; r[7] = (__bf16)b.w;
  return r;
}
static __device__ __forceinline__ v16bf cat16(v8bf x, v8bf y) {
  v16bf r;
#pragma unroll
  for (int i = 0; i < 8; ++i) { r[i] = x[i]; r[8 + i] = y[i]; }
  return r;
}
// A fragment for one 16x32 bf16 tile, from f32 row-major (scale applied)
static __device__ __forceinline__ v16bf load_afrag(const float* rowptr, float scale) {
  const float4* xp = (const float4*)rowptr;
  const float4 x0 = xp[0], x1 = xp[1], x2 = xp[4], x3 = xp[5];
  v16bf a;
  a[0] = (__bf16)(x0.x * scale); a[1] = (__bf16)(x0.y * scale);
  a[2] = (__bf16)(x0.z * scale); a[3] = (__bf16)(x0.w * scale);
  a[4] = (__bf16)(x1.x * scale); a[5] = (__bf16)(x1.y * scale);
  a[6] = (__bf16)(x1.z * scale); a[7] = (__bf16)(x1.w * scale);
  a[8] = (__bf16)(x2.x * scale); a[9] = (__bf16)(x2.y * scale);
  a[10] = (__bf16)(x2.z * scale); a[11] = (__bf16)(x2.w * scale);
  a[12] = (__bf16)(x3.x * scale); a[13] = (__bf16)(x3.y * scale);
  a[14] = (__bf16)(x3.z * scale); a[15] = (__bf16)(x3.w * scale);
  return a;
}

// ---------------------------------------------------------------------------
// Generic GEMM: Y[M,N] = X[M,K] @ W[K,N] + bias[N]   (f32 in/out, bf16 WMMA)
// Block = 256 threads (8 waves). Block tile: 128 rows x 64 cols.
// W k-tile (32x64) staged in LDS (bf16, transposed, padded row=40 -> the
// 16 col-rows touched by a half-wave map to 16 distinct bank groups),
// double buffered, one barrier per k-step.
// Requires M%128==0, N%64==0, K%32==0 (true for all layers here).
// ---------------------------------------------------------------------------
__global__ void __launch_bounds__(256)
gemm_bias_kernel(const float* __restrict__ X, const float* __restrict__ W,
                 const float* __restrict__ bias, float* __restrict__ Y,
                 int M, int K, int N) {
  constexpr int WROW = 40;                 // padded bf16 row (80B)
  __shared__ __bf16 Wt[2][64][WROW];       // [buf][col][k]
  const int tid  = threadIdx.x;
  const int lane = tid & 31;
  const int wave = tid >> 5;
  const int lo = lane & 15;
  const int hi = lane >> 4;
  const int m0 = (blockIdx.x * 8 + wave) * 16;
  const int n0 = blockIdx.y * 64;

  const int kr = tid >> 4;                 // 0..15
  const int c4 = (tid & 15) * 4;           // col group of 4

  auto stage = [&](int kk, int buf) {
#pragma unroll
    for (int p = 0; p < 2; ++p) {
      const int k = kr + p * 16;
      const float4 v = *(const float4*)(W + (size_t)(kk + k) * N + n0 + c4);
      Wt[buf][c4 + 0][k] = (__bf16)v.x;
      Wt[buf][c4 + 1][k] = (__bf16)v.y;
      Wt[buf][c4 + 2][k] = (__bf16)v.z;
      Wt[buf][c4 + 3][k] = (__bf16)v.w;
    }
  };

  stage(0, 0);
  __syncthreads();

  v8f acc[4] = {};
  const float* xrow = X + (size_t)(m0 + lo) * K;
  const int nsteps = K / 32;

  for (int s = 0; s < nsteps; ++s) {
    const int kk = s * 32;
    const int buf = s & 1;
    if (s + 1 < nsteps) stage(kk + 32, buf ^ 1);

    const v16bf a = load_afrag(xrow + kk + hi * 8, 1.0f);
#pragma unroll
    for (int t = 0; t < 4; ++t) {
      const int col = t * 16 + lo;
      const v8bf* wp = (const v8bf*)&Wt[buf][col][hi * 8];
      acc[t] = wmma_bf16(a, cat16(wp[0], wp[2]), acc[t]);
    }
    __syncthreads();
  }
  // C/D layout: row = m0 + hi*8 + r ; col = n0 + t*16 + lo
#pragma unroll
  for (int t = 0; t < 4; ++t) {
    const int col = n0 + t * 16 + lo;
    const float bv = bias[col];
#pragma unroll
    for (int r = 0; r < 8; ++r) {
      Y[(size_t)(m0 + hi * 8 + r) * N + col] = acc[t][r] + bv;
    }
  }
}

// ---------------------------------------------------------------------------
// Distances, distance mask, and distance-bias MLP (dist1->relu->dist2).
// One thread per (b,s,a) pair. distances & mask -> d_out directly.
// dbias stored as [B,H,S,A] f32 so attention reads are lane-contiguous.
// ---------------------------------------------------------------------------
__global__ void __launch_bounds__(256)
dist_bias_kernel(const float* __restrict__ pc, const float* __restrict__ lc,
                 const float* __restrict__ w1, const float* __restrict__ b1,
                 const float* __restrict__ w2, const float* __restrict__ b2,
                 float* __restrict__ out_dist, float* __restrict__ out_mask,
                 float* __restrict__ dbias) {
  __shared__ float sw1[32], sb1[32], sw2[32 * 8], sb2[8];
  const int t = threadIdx.x;
  if (t < 32) { sw1[t] = w1[t]; sb1[t] = b1[t]; }
  if (t < 8)  { sb2[t] = b2[t]; }
  sw2[t] = w2[t];                    // block has exactly 256 threads
  __syncthreads();

  const size_t idx = (size_t)blockIdx.x * 256 + t;   // (b*S+s)*A + a
  const int a = (int)(idx % Ac);
  const size_t bs = idx / Ac;                        // b*S + s
  const int b = (int)(bs / Sc);

  const float* p = pc + bs * 3;
  const float* l = lc + ((size_t)b * Ac + a) * 3;
  const float dx = p[0] - l[0], dy = p[1] - l[1], dz = p[2] - l[2];
  const float d = sqrtf(dx * dx + dy * dy + dz * dz);

  out_dist[idx] = d;
  out_mask[idx] = (d <= 10.0f) ? 1.0f : 0.0f;

  float acc[8];
#pragma unroll
  for (int hh = 0; hh < 8; ++hh) acc[hh] = sb2[hh];
#pragma unroll 4
  for (int j = 0; j < 32; ++j) {
    const float hv = fmaxf(fmaf(d, sw1[j], sb1[j]), 0.0f);
#pragma unroll
    for (int hh = 0; hh < 8; ++hh) acc[hh] = fmaf(hv, sw2[j * 8 + hh], acc[hh]);
  }
  // [B,H,S,A]: target = b*H*S*A + hh*S*A + s*A + a = idx + b*(H-1)*plane + hh*plane
  const size_t plane = (size_t)Sc * Ac;
  float* db = dbias + idx + (size_t)b * (Hc - 1) * plane;
#pragma unroll
  for (int hh = 0; hh < 8; ++hh) db[(size_t)hh * plane] = acc[hh];
}

// ---------------------------------------------------------------------------
// protein -> ligand attention. A = 128 keys: whole score row fits in regs.
// Block = 256 (8 waves, all same (b,h)); wave handles one 16-query tile.
// K (row-major) and V (transposed) staged once in LDS as bf16.
// ---------------------------------------------------------------------------
__global__ void __launch_bounds__(256)
attn_p2l_kernel(const float* __restrict__ Qm, const float* __restrict__ Km,
                const float* __restrict__ Vm, const float* __restrict__ dbias,
                const float* __restrict__ dist, const unsigned char* __restrict__ lmask,
                float* __restrict__ Out) {
  __shared__ __bf16 Kt[128][40];        // [key][hd]   (padded)
  __shared__ __bf16 Vt[32][136];        // [hd][key]   (padded)
  __shared__ __bf16 Plds[8][16][136];   // per-wave probabilities (padded)
  const int tid  = threadIdx.x;
  const int lane = tid & 31;
  const int wave = tid >> 5;
  const int lo = lane & 15, hi = lane >> 4;
  const int st = blockIdx.x * 8 + wave;      // query tile over S
  const int h = blockIdx.y, b = blockIdx.z;
  const int s0 = st * 16;
  const float scale = 0.17677669529663687f;  // 1/sqrt(HD=32)

  // ---- stage K and V tiles (shared by all waves of this block) ----
  {
    const int key  = tid >> 1;
    const int half = (tid & 1) * 16;
    const float4* kp = (const float4*)(Km + (size_t)(b * Ac + key) * Dc + h * 32 + half);
    *(v8bf*)&Kt[key][half]     = pack8(kp[0], kp[1]);
    *(v8bf*)&Kt[key][half + 8] = pack8(kp[2], kp[3]);
    const float4* vp = (const float4*)(Vm + (size_t)(b * Ac + key) * Dc + h * 32 + half);
    float vv[16];
    *(float4*)&vv[0]  = vp[0];
    *(float4*)&vv[4]  = vp[1];
    *(float4*)&vv[8]  = vp[2];
    *(float4*)&vv[12] = vp[3];
#pragma unroll
    for (int j = 0; j < 16; ++j) Vt[half + j][key] = (__bf16)vv[j];
  }
  __syncthreads();

  // ---- Q fragment (scale folded in) ----
  const v16bf qa =
      load_afrag(Qm + (size_t)(b * Sc + s0 + lo) * Dc + h * 32 + hi * 8, scale);

  // ---- scores: 8 key tiles of 16 ----
  v8f sc[8];
#pragma unroll
  for (int kt = 0; kt < 8; ++kt) {
    const int key = kt * 16 + lo;
    const v8bf* kp = (const v8bf*)&Kt[key][hi * 8];
    v8f z = {};
    sc[kt] = wmma_bf16(qa, cat16(kp[0], kp[2]), z);
  }

  // ---- bias + mask ----
#pragma unroll
  for (int kt = 0; kt < 8; ++kt) {
    const int acol = kt * 16 + lo;
    const bool lm = lmask[b * Ac + acol] != 0;
#pragma unroll
    for (int r = 0; r < 8; ++r) {
      const int srow = s0 + hi * 8 + r;
      const size_t didx = (size_t)(b * Sc + srow) * Ac + acol;
      const size_t bidx = (size_t)((b * Hc + h) * Sc + srow) * Ac + acol;
      const float v = sc[kt][r] + dbias[bidx];
      const bool ok = lm && (dist[didx] <= 10.0f);
      sc[kt][r] = ok ? v : -__builtin_inff();
    }
  }

  // ---- row softmax (row r lives in a half-wave; xor 1..8 stay in-half) ----
  float rinv[8];
#pragma unroll
  for (int r = 0; r < 8; ++r) {
    float m = sc[0][r];
#pragma unroll
    for (int kt = 1; kt < 8; ++kt) m = fmaxf(m, sc[kt][r]);
    m = fmaxf(m, __shfl_xor(m, 1, 32));
    m = fmaxf(m, __shfl_xor(m, 2, 32));
    m = fmaxf(m, __shfl_xor(m, 4, 32));
    m = fmaxf(m, __shfl_xor(m, 8, 32));
    float s = 0.f;
#pragma unroll
    for (int kt = 0; kt < 8; ++kt) {
      const float p = expf(sc[kt][r] - m);
      sc[kt][r] = p;
      s += p;
    }
    s += __shfl_xor(s, 1, 32);
    s += __shfl_xor(s, 2, 32);
    s += __shfl_xor(s, 4, 32);
    s += __shfl_xor(s, 8, 32);
    rinv[r] = 1.0f / s;
  }

  // ---- stage normalized P in LDS (bf16) ----
#pragma unroll
  for (int kt = 0; kt < 8; ++kt)
#pragma unroll
    for (int r = 0; r < 8; ++r)
      Plds[wave][hi * 8 + r][kt * 16 + lo] = (__bf16)(sc[kt][r] * rinv[r]);
  __syncthreads();

  // ---- O = P @ V  (4 key-chunks of 32, 2 output col tiles) ----
  v8f o0 = {}, o1 = {};
#pragma unroll
  for (int c = 0; c < 4; ++c) {
    const v8bf p0 = *(const v8bf*)&Plds[wave][lo][c * 32 + hi * 8];
    const v8bf p1 = *(const v8bf*)&Plds[wave][lo][c * 32 + hi * 8 + 16];
    const v16bf pa = cat16(p0, p1);
#pragma unroll
    for (int nt = 0; nt < 2; ++nt) {
      const int hd = nt * 16 + lo;
      const v8bf* vp = (const v8bf*)&Vt[hd][c * 32 + hi * 8];
      const v16bf vf = cat16(vp[0], vp[2]);
      if (nt == 0) o0 = wmma_bf16(pa, vf, o0);
      else         o1 = wmma_bf16(pa, vf, o1);
    }
  }
#pragma unroll
  for (int r = 0; r < 8; ++r) {
    const size_t rowoff = (size_t)(b * Sc + s0 + hi * 8 + r) * Dc + h * 32 + lo;
    Out[rowoff]      = o0[r];
    Out[rowoff + 16] = o1[r];
  }
}

// ---------------------------------------------------------------------------
// ligand -> protein attention. S = 2048 keys: flash-style online softmax,
// 32 keys per iteration. One wave per block (spread across WGPs).
// ---------------------------------------------------------------------------
__global__ void __launch_bounds__(32)
attn_l2p_kernel(const float* __restrict__ Qm, const float* __restrict__ Km,
                const float* __restrict__ Vm, const float* __restrict__ dbias,
                const float* __restrict__ dist, const unsigned char* __restrict__ pmask,
                float* __restrict__ Out) {
  __shared__ __bf16 Plds[16][40];     // padded: conflict-free b128 reads
  const int lane = threadIdx.x & 31;
  const int lo = lane & 15, hi = lane >> 4;
  const int qt = blockIdx.x, h = blockIdx.y, b = blockIdx.z;
  const int a0 = qt * 16;
  const float scale = 0.17677669529663687f;

  const v16bf qa =
      load_afrag(Qm + (size_t)(b * Ac + a0 + lo) * Dc + h * 32 + hi * 8, scale);

  float rmax[8], rsum[8];
#pragma unroll
  for (int r = 0; r < 8; ++r) { rmax[r] = -__builtin_inff(); rsum[r] = 0.f; }
  v8f o0 = {}, o1 = {};

  for (int k0 = 0; k0 < Sc; k0 += 32) {
    v8f stt[2];
#pragma unroll
    for (int t = 0; t < 2; ++t) {
      const int key = k0 + t * 16 + lo;
      // K fragment: 8 f32 at hi*8 and 8 f32 at hi*8+16 -> 4x b128
      const float4* kp = (const float4*)(Km + (size_t)(b * Sc + key) * Dc + h * 32 + hi * 8);
      v8f z = {};
      stt[t] = wmma_bf16(qa, cat16(pack8(kp[0], kp[1]), pack8(kp[4], kp[5])), z);
      const bool pm = pmask[(size_t)b * Sc + key] != 0;
#pragma unroll
      for (int r = 0; r < 8; ++r) {
        const int arow = a0 + hi * 8 + r;
        const size_t didx = (size_t)(b * Sc + key) * Ac + arow;
        const size_t bidx = (size_t)((b * Hc + h) * Sc + key) * Ac + arow;
        const float v = stt[t][r] + dbias[bidx];
        const bool ok = pm && (dist[didx] <= 10.0f);
        stt[t][r] = ok ? v : -__builtin_inff();
      }
    }
    // online softmax update (per row, replicated across half-wave)
#pragma unroll
    for (int r = 0; r < 8; ++r) {
      float tm = fmaxf(stt[0][r], stt[1][r]);
      tm = fmaxf(tm, __shfl_xor(tm, 1, 32));
      tm = fmaxf(tm, __shfl_xor(tm, 2, 32));
      tm = fmaxf(tm, __shfl_xor(tm, 4, 32));
      tm = fmaxf(tm, __shfl_xor(tm, 8, 32));
      const float nm = fmaxf(rmax[r], tm);
      const bool dead = (nm == -__builtin_inff());
      const float scl = dead ? 1.0f : expf(rmax[r] - nm);
      const float p0 = dead ? 0.0f : expf(stt[0][r] - nm);
      const float p1 = dead ? 0.0f : expf(stt[1][r] - nm);
      stt[0][r] = p0;
      stt[1][r] = p1;
      float ps = p0 + p1;
      ps += __shfl_xor(ps, 1, 32);
      ps += __shfl_xor(ps, 2, 32);
      ps += __shfl_xor(ps, 4, 32);
      ps += __shfl_xor(ps, 8, 32);
      rsum[r] = rsum[r] * scl + ps;
      rmax[r] = nm;
      o0[r] *= scl;
      o1[r] *= scl;
    }
    // stage P (16x32) in LDS; single wave: in-order LDS gives RAW safety
#pragma unroll
    for (int t = 0; t < 2; ++t)
#pragma unroll
      for (int r = 0; r < 8; ++r)
        Plds[hi * 8 + r][t * 16 + lo] = (__bf16)stt[t][r];

    const v8bf p0v = *(const v8bf*)&Plds[lo][hi * 8];
    const v8bf p1v = *(const v8bf*)&Plds[lo][hi * 8 + 16];
    const v16bf pa = cat16(p0v, p1v);
#pragma unroll
    for (int nt = 0; nt < 2; ++nt) {
      const int col = h * 32 + nt * 16 + lo;
      v16bf vf;
#pragma unroll
      for (int i = 0; i < 16; ++i) {
        const int key = k0 + hi * 8 + (i & 7) + (i >> 3) * 16;
        vf[i] = (__bf16)Vm[(size_t)(b * Sc + key) * Dc + col];
      }
      if (nt == 0) o0 = wmma_bf16(pa, vf, o0);
      else         o1 = wmma_bf16(pa, vf, o1);
    }
  }
#pragma unroll
  for (int r = 0; r < 8; ++r) {
    const float inv = 1.0f / rsum[r];
    const size_t rowoff = (size_t)(b * Ac + a0 + hi * 8 + r) * Dc + h * 32 + lo;
    Out[rowoff]      = o0[r] * inv;
    Out[rowoff + 16] = o1[r] * inv;
  }
}

// ---------------------------------------------------------------------------
// Y[row] = LayerNorm(X0[row] + Xl[row]) * g + b   (wave per row, float4)
// Wd % 128 == 0 (384, 128)
// ---------------------------------------------------------------------------
__global__ void __launch_bounds__(256)
resid_ln_kernel(const float* __restrict__ X0, const float* __restrict__ Xl,
                const float* __restrict__ g, const float* __restrict__ bb,
                float* __restrict__ Y, int rows, int Wd) {
  const int lane = threadIdx.x & 31;
  const int wave = threadIdx.x >> 5;
  const int row = blockIdx.x * 8 + wave;
  if (row >= rows) return;
  const float4* x04 = (const float4*)(X0 + (size_t)row * Wd);
  const float4* xl4 = (const float4*)(Xl + (size_t)row * Wd);
  const int nw4 = Wd >> 2;
  float s = 0.f, ss = 0.f;
  for (int i = lane; i < nw4; i += 32) {
    const float4 a = x04[i], c = xl4[i];
    const float v0 = a.x + c.x, v1 = a.y + c.y, v2 = a.z + c.z, v3 = a.w + c.w;
    s += v0 + v1 + v2 + v3;
    ss += v0 * v0 + v1 * v1 + v2 * v2 + v3 * v3;
  }
#pragma unroll
  for (int m = 1; m < 32; m <<= 1) {
    s  += __shfl_xor(s, m, 32);
    ss += __shfl_xor(ss, m, 32);
  }
  const float mu = s / (float)Wd;
  const float var = ss / (float)Wd - mu * mu;
  const float rstd = rsqrtf(var + 1e-5f);
  float4* y4 = (float4*)(Y + (size_t)row * Wd);
  const float4* g4 = (const float4*)g;
  const float4* b4 = (const float4*)bb;
  for (int i = lane; i < nw4; i += 32) {
    const float4 a = x04[i], c = xl4[i], gg = g4[i], bv = b4[i];
    float4 o;
    o.x = (a.x + c.x - mu) * rstd * gg.x + bv.x;
    o.y = (a.y + c.y - mu) * rstd * gg.y + bv.y;
    o.z = (a.z + c.z - mu) * rstd * gg.z + bv.z;
    o.w = (a.w + c.w - mu) * rstd * gg.w + bv.w;
    y4[i] = o;
  }
}

// ---------------------------------------------------------------------------
// Host-side orchestration
// ---------------------------------------------------------------------------
extern "C" void kernel_launch(void* const* d_in, const int* in_sizes, int n_in,
                              void* d_out, int out_size, void* d_ws, size_t ws_size,
                              hipStream_t stream) {
  (void)in_sizes; (void)n_in; (void)out_size; (void)ws_size;
  const float* pf     = (const float*)d_in[0];
  const float* lfeat  = (const float*)d_in[1];
  const float* pcoord = (const float*)d_in[2];
  const float* lcoord = (const float*)d_in[3];
  const unsigned char* pmask = (const unsigned char*)d_in[4];
  const unsigned char* lmask = (const unsigned char*)d_in[5];
  const float* Wpp = (const float*)d_in[6];  const float* bpp = (const float*)d_in[7];
  const float* Wlp = (const float*)d_in[8];  const float* blp = (const float*)d_in[9];
  const float* Wpq = (const float*)d_in[10]; const float* bpq = (const float*)d_in[11];
  const float* Wlk = (const float*)d_in[12]; const float* blkb = (const float*)d_in[13];
  const float* Wlv = (const float*)d_in[14]; const float* blv = (const float*)d_in[15];
  const float* Wlq = (const float*)d_in[16]; const float* blq = (const float*)d_in[17];
  const float* Wpk = (const float*)d_in[18]; const float* bpk = (const float*)d_in[19];
  const float* Wpv = (const float*)d_in[20]; const float* bpv = (const float*)d_in[21];
  const float* Wd1 = (const float*)d_in[22]; const float* bd1 = (const float*)d_in[23];
  const float* Wd2 = (const float*)d_in[24]; const float* bd2 = (const float*)d_in[25];
  const float* Wpo = (const float*)d_in[26]; const float* bpo = (const float*)d_in[27];
  const float* Wlo = (const float*)d_in[28]; const float* blo = (const float*)d_in[29];
  const float* png = (const float*)d_in[30]; const float* pnb = (const float*)d_in[31];
  const float* lng = (const float*)d_in[32]; const float* lnb = (const float*)d_in[33];

  const size_t MP = (size_t)Bc * Sc;   // 16384 protein rows
  const size_t ML = (size_t)Bc * Ac;   // 1024 ligand rows

  float* ws = (float*)d_ws;
  size_t off = 0;
  auto take = [&](size_t n) { float* p = ws + off; off += n; return p; };
  float* p_proj = take(MP * Dc);
  float* l_proj = take(ML * Dc);
  float* q1     = take(MP * Dc);
  float* k2     = take(MP * Dc);
  float* v2     = take(MP * Dc);
  float* kl     = take(ML * Dc);
  float* vl     = take(ML * Dc);
  float* ql     = take(ML * Dc);
  float* p_att  = take(MP * Dc);
  float* l_att  = take(ML * Dc);
  float* p_lin  = take(MP * PDc);
  float* l_lin  = take(ML * LDc);
  float* dbias  = take((size_t)Bc * Hc * Sc * Ac);   // [B,H,S,A]

  float* out   = (float*)d_out;
  float* p_out = out;
  float* l_out = p_out + MP * PDc;
  float* dists = l_out + ML * LDc;
  float* maskf = dists + (size_t)Bc * Sc * Ac;

  const dim3 tb(256);

  // Projections
  gemm_bias_kernel<<<dim3(128, 4), tb, 0, stream>>>(pf,    Wpp, bpp, p_proj, 16384, PDc, Dc);
  gemm_bias_kernel<<<dim3(8, 4),   tb, 0, stream>>>(lfeat, Wlp, blp, l_proj, 1024,  LDc, Dc);

  // Distances + distance-bias MLP (distances/mask straight to d_out)
  dist_bias_kernel<<<dim3(8192), tb, 0, stream>>>(pcoord, lcoord, Wd1, bd1, Wd2, bd2,
                                                  dists, maskf, dbias);

  // Q/K/V linears
  gemm_bias_kernel<<<dim3(128, 4), tb, 0, stream>>>(p_proj, Wpq, bpq, q1, 16384, Dc, Dc);
  gemm_bias_kernel<<<dim3(8, 4),   tb, 0, stream>>>(l_proj, Wlk, blkb, kl, 1024, Dc, Dc);
  gemm_bias_kernel<<<dim3(8, 4),   tb, 0, stream>>>(l_proj, Wlv, blv, vl, 1024, Dc, Dc);
  gemm_bias_kernel<<<dim3(8, 4),   tb, 0, stream>>>(l_proj, Wlq, blq, ql, 1024, Dc, Dc);
  gemm_bias_kernel<<<dim3(128, 4), tb, 0, stream>>>(p_proj, Wpk, bpk, k2, 16384, Dc, Dc);
  gemm_bias_kernel<<<dim3(128, 4), tb, 0, stream>>>(p_proj, Wpv, bpv, v2, 16384, Dc, Dc);

  // Cross attention (both directions)
  attn_p2l_kernel<<<dim3(16, Hc, Bc), tb, 0, stream>>>(q1, kl, vl, dbias, dists, lmask, p_att);
  attn_l2p_kernel<<<dim3(8, Hc, Bc), dim3(32), 0, stream>>>(ql, k2, v2, dbias, dists, pmask, l_att);

  // Output linears + residual layernorm
  gemm_bias_kernel<<<dim3(128, 6), tb, 0, stream>>>(p_att, Wpo, bpo, p_lin, 16384, Dc, PDc);
  gemm_bias_kernel<<<dim3(8, 2),   tb, 0, stream>>>(l_att, Wlo, blo, l_lin, 1024,  Dc, LDc);
  resid_ln_kernel<<<dim3(2048), tb, 0, stream>>>(pf,    p_lin, png, pnb, p_out, 16384, PDc);
  resid_ln_kernel<<<dim3(128),  tb, 0, stream>>>(lfeat, l_lin, lng, lnb, l_out, 1024,  LDc);
}
